// SelfAttention_28561532519037
// MI455X (gfx1250) — matrix operand order repaired and checked by reading, not compiled, not run on hardware
//
#include <hip/hip_runtime.h>
#include <hip/hip_bf16.h>
#include <math.h>

// ---------------- fixed problem geometry (from reference setup) ------------
#define N_     64
#define C_     256
#define T_     128
#define V_     25
#define E2_    512      // 2 * h * d
#define H_     8
#define D_     32
#define G_     2                // (n,t) pairs per workgroup (W amortization)
#define NB_    (N_ * T_ / G_)   // 4096 workgroups
#define NTILES (E2_ / 16)       // 32 N-tiles in projection
#define KTILES (C_ / 32)        // 8 K-steps of 32
#define MROWS  (G_ * 32)        // 64 padded A rows per workgroup
#define XTILE_BYTES (C_ * V_ * 4)   // 25600 B per (n,t) x tile in LDS

typedef __attribute__((ext_vector_type(16))) __bf16  v16bf;
typedef __attribute__((ext_vector_type(8)))  float   v8f;
typedef __attribute__((ext_vector_type(4)))  unsigned int uint4x;
typedef __attribute__((ext_vector_type(4)))  unsigned int u32x4;
typedef __attribute__((ext_vector_type(8)))  int i32x8;
typedef __attribute__((ext_vector_type(4)))  int i32x4;

union Frag {            // one WMMA 16-bit A/B operand: 16 bf16 = 8 VGPRs
    v16bf  v;
    uint4x q[2];
};

__device__ __forceinline__ unsigned short f2bf(float f) {
    unsigned u = __builtin_bit_cast(unsigned, f);
    unsigned r = u + 0x7FFFu + ((u >> 16) & 1u);   // round-to-nearest-even
    return (unsigned short)(r >> 16);
}

// ---------------------------------------------------------------------------
// Pack W (512x256 fp32, row-major [e][c]) into bf16 WMMA B-fragment order:
// each lane of a wave loads its whole B fragment as one contiguous 32B.
// ---------------------------------------------------------------------------
__global__ void pack_w_kernel(const float* __restrict__ W,
                              unsigned short* __restrict__ wpack) {
    int idx = blockIdx.x * blockDim.x + threadIdx.x;     // 0 .. 8191
    if (idx >= NTILES * KTILES * 32) return;
    int lane  = idx & 31;
    int kt    = (idx >> 5) & 7;
    int ntile = idx >> 8;
    int lo = lane & 15, hi = lane >> 4;
    int e = ntile * 16 + lo;

    __align__(16) unsigned short vals[16];
#pragma unroll
    for (int j = 0; j < 8; ++j) {
#pragma unroll
        for (int p = 0; p < 2; ++p) {
            int c = kt * 32 + 16 * hi + 2 * j + p;       // B[k=c][n=e]
            vals[2 * j + p] = f2bf(W[e * C_ + c]);
        }
    }
    uint4x* dst = (uint4x*)(wpack + (size_t)idx * 16);
    dst[0] = *(const uint4x*)&vals[0];
    dst[1] = *(const uint4x*)&vals[8];
}

// ---------------------------------------------------------------------------
// Fused LayerNorm + QK projection + per-head attention softmax.
// One workgroup = two (n,t) pairs; 256 threads = 8 wave32.
// LDS (96KB total, <= 320KB/WGP so 2 WGs can co-reside):
//   region0 (64KB) : two x tiles (fp32 [256][25], TDM-loaded) reused as
//                    qk bf16[64][512]
//   region1 (32KB) : normalized A tile bf16[64][256]
// ---------------------------------------------------------------------------
__global__ __launch_bounds__(256)
void fused_attn_kernel(const float* __restrict__ x,
                       const float* __restrict__ gamma,
                       const float* __restrict__ beta,
                       const float* __restrict__ bias,
                       const unsigned short* __restrict__ wpack,
                       float* __restrict__ out) {
    __shared__ __align__(16) unsigned char smem[MROWS * E2_ * 2 + MROWS * C_ * 2];
    unsigned short (*qk)[E2_] = (unsigned short(*)[E2_])smem;                    // 64KB
    unsigned short (*Al)[C_]  = (unsigned short(*)[C_])(smem + MROWS * E2_ * 2); // 32KB

    const int tid  = threadIdx.x;
    const int wave = tid >> 5;
    const int lane = tid & 31;
    const int lo   = lane & 15;
    const int hi   = lane >> 4;
    const int bi0  = blockIdx.x * G_;     // first flattened (n,t) index

    // ---------------- phase 1: TDM async tensor load of both x tiles ------
    // x[n][c][t][v]: tile for (n,t) = 256 rows (c) x 25 contiguous fp32,
    // row stride 3200 elements. One tensor_load_to_lds per tile, issued by
    // wave 0 only (TDM ignores EXEC; whole-wave branch gates issue).
    if (wave == 0) {
#pragma unroll
        for (int g = 0; g < G_; ++g) {
            int bi  = bi0 + g;
            int n_i = bi / T_;
            int t_i = bi - n_i * T_;
            unsigned long long gaddr = (unsigned long long)(uintptr_t)x +
                4ull * ((unsigned long long)n_i * (C_ * T_ * V_) +
                        (unsigned long long)t_i * V_);
            unsigned lds_off = (unsigned)(g * XTILE_BYTES);

            u32x4 g0;
            g0.x = 1u;                                    // count=1, user mode
            g0.y = lds_off;                               // LDS byte address
            g0.z = (unsigned)(gaddr & 0xffffffffull);     // global_addr[31:0]
            g0.w = (unsigned)(gaddr >> 32) | 0x80000000u; // [56:32] | type=2

            i32x8 g1;
            g1.s0 = 0x00020000;          // workgroup_mask=0, data_size=2 (4B)
            g1.s1 = (int)(((unsigned)(T_ * V_) & 0xffffu) << 16); // tensor_dim0[15:0]
            g1.s2 = (int)(((unsigned)C_ & 0xffffu) << 16);        // tensor_dim1[15:0]
            g1.s3 = (int)(((unsigned)V_) << 16);                  // tile_dim0 = 25
            g1.s4 = (int)C_;                                      // tile_dim1 = 256
            g1.s5 = (int)(T_ * V_);                               // dim0_stride = 3200
            g1.s6 = 0;
            g1.s7 = 0;

            i32x4 gz  = {0, 0, 0, 0};            // groups 2/3: unused (2D tile)
            i32x8 gz8 = {0, 0, 0, 0, 0, 0, 0, 0};
            __builtin_amdgcn_tensor_load_to_lds(g0, g1, gz, gz, gz8, 0);
        }
        __builtin_amdgcn_s_wait_tensorcnt(0);
    }
    __syncthreads();

    // ---------------- phase 2: LayerNorm over c, per (g,v) column ---------
    // xs tile g: float[256][25] at smem + g*25600 (stride 25 is odd ->
    // lane-strided column reads are bank-conflict free).
    for (int idx = wave; idx < G_ * V_; idx += 8) {
        int g  = idx / V_;
        int vi = idx - g * V_;
        const float* xs = (const float*)(smem + g * XTILE_BYTES);
        float s = 0.f;
        for (int c = lane; c < C_; c += 32) s += xs[c * V_ + vi];
#pragma unroll
        for (int off = 16; off > 0; off >>= 1) s += __shfl_xor(s, off, 32);
        float mu = s * (1.f / C_);
        float vs = 0.f;
        for (int c = lane; c < C_; c += 32) { float d = xs[c * V_ + vi] - mu; vs += d * d; }
#pragma unroll
        for (int off = 16; off > 0; off >>= 1) vs += __shfl_xor(vs, off, 32);
        float rstd = rsqrtf(vs * (1.f / C_) + 1e-5f);
        for (int c = lane; c < C_; c += 32) {
            float val = (xs[c * V_ + vi] - mu) * rstd * gamma[c] + beta[c];
            Al[g * 32 + vi][c] = f2bf(val);
        }
    }
    // zero pad rows v = 25..31 of each group so WMMA accumulates cleanly
    for (int i = tid; i < G_ * (32 - V_) * C_; i += 256) {
        int r = i >> 8;                 // 0 .. 13
        int g = r / (32 - V_);
        int rr = V_ + r - g * (32 - V_);
        Al[g * 32 + rr][i & (C_ - 1)] = 0;
    }
    __syncthreads();

    // ---------------- phase 3: projection GEMM (bf16 WMMA) ----------------
    // A: [64 x 256] in Al ; B: prepacked W ; C: qk bf16 [64 x 512] (+bias)
    // One B fragment feeds 4 M-tiles -> 4x W-reuse vs. unblocked version.
    const uint4x* wp4 = (const uint4x*)wpack;
    for (int nt4 = 0; nt4 < NTILES / 8; ++nt4) {          // 4 N-tiles per wave
        int ntile = wave * 4 + nt4;
        v8f acc[4] = {};
        float bval = bias[ntile * 16 + lo];
#pragma unroll
        for (int kt = 0; kt < KTILES; ++kt) {
            Frag bf_;
            int fidx = (ntile * KTILES + kt) * 32 + lane;
            bf_.q[0] = wp4[fidx * 2];
            bf_.q[1] = wp4[fidx * 2 + 1];
#pragma unroll
            for (int mt = 0; mt < 4; ++mt) {
                Frag a;
                // A k order used here (contiguous, not the ISA table order): VGPR0-7 <- K = 16*hi..16*hi+15 (matches packed B)
                const uint4x* p = (const uint4x*)&Al[16 * mt + lo][kt * 32 + 16 * hi];
                a.q[0] = p[0]; a.q[1] = p[1];
                acc[mt] = __builtin_amdgcn_wmma_f32_16x16x32_bf16(
                              false, a.v, false, bf_.v, (short)0, acc[mt], false, false);
            }
        }
        // C layout: VGPR r -> row (r + 8*hi), col = ntile*16 + lo
        int e = ntile * 16 + lo;
#pragma unroll
        for (int mt = 0; mt < 4; ++mt)
#pragma unroll
            for (int r = 0; r < 8; ++r)
                qk[16 * mt + r + 8 * hi][e] = f2bf(acc[mt][r] + bval);
    }
    __syncthreads();

    // ---------------- phase 4: per-head attention (2 problems per wave) ---
    const float scale = 0.17677669529663687f;  // 32^-0.5
#pragma unroll
    for (int it = 0; it < 2; ++it) {
        int idx = wave + it * 8;        // 0..15
        int g = idx >> 3;               // group (which (n,t) pair)
        int h = idx & 7;                // head
        int rbase = g * 32;

        Frag qa[2], kb[2];
#pragma unroll
        for (int mt = 0; mt < 2; ++mt) {   // q rows as A-matrix, K = d = 32
            const uint4x* p = (const uint4x*)&qk[rbase + 16 * mt + lo][32 * h + 16 * hi];
            qa[mt].q[0] = p[0]; qa[mt].q[1] = p[1];
        }
#pragma unroll
        for (int nt = 0; nt < 2; ++nt) {   // k rows as B columns: B[k=d][n=j]
            const uint4x* p = (const uint4x*)&qk[rbase + 16 * nt + lo][256 + 32 * h + 16 * hi];
            kb[nt].q[0] = p[0]; kb[nt].q[1] = p[1];
        }
        v8f d00 = {}, d01 = {}, d10 = {}, d11 = {};
        d00 = __builtin_amdgcn_wmma_f32_16x16x32_bf16(false, qa[0].v, false, kb[0].v, (short)0, d00, false, false);
        d01 = __builtin_amdgcn_wmma_f32_16x16x32_bf16(false, qa[0].v, false, kb[1].v, (short)0, d01, false, false);
        d10 = __builtin_amdgcn_wmma_f32_16x16x32_bf16(false, qa[1].v, false, kb[0].v, (short)0, d10, false, false);
        d11 = __builtin_amdgcn_wmma_f32_16x16x32_bf16(false, qa[1].v, false, kb[1].v, (short)0, d11, false, false);

        float* outp = out + ((size_t)(bi0 + g) * H_ + h) * (V_ * V_);
#pragma unroll
        for (int mt = 0; mt < 2; ++mt) {
            v8f cA = mt ? d10 : d00;               // cols  0..15
            v8f cB = mt ? d11 : d01;               // cols 16..31 (valid < 25)
#pragma unroll
            for (int r = 0; r < 8; ++r) {
                int i = 16 * mt + r + 8 * hi;      // dots row (query index)
                float v0 = cA[r] * scale;
                float v1 = (lo < 9) ? cB[r] * scale : -3.0e38f;
                float m = fmaxf(v0, v1);
                m = fmaxf(m, __shfl_xor(m, 1, 16));
                m = fmaxf(m, __shfl_xor(m, 2, 16));
                m = fmaxf(m, __shfl_xor(m, 4, 16));
                m = fmaxf(m, __shfl_xor(m, 8, 16));
                float e0 = __expf(v0 - m);
                float e1 = (lo < 9) ? __expf(v1 - m) : 0.f;
                float s = e0 + e1;
                s += __shfl_xor(s, 1, 16);
                s += __shfl_xor(s, 2, 16);
                s += __shfl_xor(s, 4, 16);
                s += __shfl_xor(s, 8, 16);
                float inv = 1.f / s;
                if (i < V_) {
                    outp[i * V_ + lo] = e0 * inv;
                    if (lo < 9) outp[i * V_ + 16 + lo] = e1 * inv;
                }
            }
        }
    }
}

// ---------------------------------------------------------------------------
extern "C" void kernel_launch(void* const* d_in, const int* in_sizes, int n_in,
                              void* d_out, int out_size, void* d_ws, size_t ws_size,
                              hipStream_t stream) {
    const float* x     = (const float*)d_in[0];   // (64,256,128,25)
    const float* gamma = (const float*)d_in[1];   // (256,)
    const float* beta  = (const float*)d_in[2];   // (256,)
    const float* W     = (const float*)d_in[3];   // (512,256)
    const float* b     = (const float*)d_in[4];   // (512,)
    // d_in[5] = n_heads (=8, hardcoded)
    unsigned short* wpack = (unsigned short*)d_ws;   // 512*256*2 = 256KB
    float* out = (float*)d_out;                      // (8192, 8, 25, 25)

    pack_w_kernel<<<32, 256, 0, stream>>>(W, wpack);
    fused_attn_kernel<<<NB_, 256, 0, stream>>>(x, gamma, beta, b, wpack, out);
}
